// FHT2D_37297495998889
// MI455X (gfx1250) — matrix operand order repaired and checked by reading, not compiled, not run on hardware
//
#include <hip/hip_runtime.h>

// ---------------------------------------------------------------------------
// 2D FHT = two batched 512x512x512 GEMMs (48 batches).
// f32 emulated as bf16 hi/lo split: A*B ~= Ah*Bh + Ah*Bl + Al*Bh
// (3x v_wmma_f32_16x16x32_bf16, f32 accumulate).
// Pre-pass splits E_col (transposed) / E_row once; GEMM-1 writes t as split
// bf16 transposed, so all GEMM-2 staging is raw copies done with
// global_load_async_to_lds_b128 (ASYNCcnt), double-buffered LDS with a
// SINGLE barrier per k-step. freqshift + 1/(H*W) folded into index mapping.
// ---------------------------------------------------------------------------

typedef __bf16 v16bf __attribute__((ext_vector_type(16)));
typedef __bf16 v8bf  __attribute__((ext_vector_type(8)));
typedef float  v8f   __attribute__((ext_vector_type(8)));
typedef float  v4f   __attribute__((ext_vector_type(4)));

#define NN     512
#define TILE   128          // output tile per 256-thread block
#define TK     32           // K step (bf16 WMMA K)
#define PITCH  40           // LDS row pitch in bf16 elems (pad 32->40, mult of 8)
#define KSTEPS (NN / TK)    // 16

__device__ __forceinline__ unsigned short bf16_rne(float f) {
    unsigned u = __float_as_uint(f);
    return (unsigned short)((u + 0x7FFFu + ((u >> 16) & 1u)) >> 16);
}
// hi = truncation (exact top 16 bits), lo = RNE of remainder (~2^-17 residual)
__device__ __forceinline__ void splitf(float f, unsigned short& h, unsigned short& l) {
    unsigned u = __float_as_uint(f);
    h = (unsigned short)(u >> 16);
    l = bf16_rne(f - __uint_as_float(u & 0xffff0000u));
}

// -------- CDNA5 async global->LDS copy (tracked on ASYNCcnt) --------
__device__ __forceinline__ void async_b128(unsigned lds_addr, const void* base,
                                           unsigned byte_off) {
    asm volatile("global_load_async_to_lds_b128 %0, %1, %2"
                 :: "v"(lds_addr), "v"(byte_off), "s"(base)
                 : "memory");
}
__device__ __forceinline__ void wait_async0() {
    asm volatile("s_wait_asynccnt 0" ::: "memory");
}

// ---------------------------------------------------------------------------
// Pre-pass: split E matrices to bf16 hi/lo. E_col stored TRANSPOSED [n][w]
// (so GEMM B tiles are straight row copies); E_row stored row-major [m][h].
// ---------------------------------------------------------------------------
__global__ __launch_bounds__(256)
void fht2d_split_E(const float* __restrict__ E_col, const float* __restrict__ E_row,
                   unsigned short* __restrict__ EcT_hi, unsigned short* __restrict__ EcT_lo,
                   unsigned short* __restrict__ Er_hi,  unsigned short* __restrict__ Er_lo)
{
    int i = blockIdx.x * 256 + threadIdx.x;     // 0 .. 512*512-1
    int r = i >> 9, c = i & (NN - 1);           // i = r*512 + c
    unsigned short h, l;
    splitf(E_col[i], h, l);
    EcT_hi[c * NN + r] = h;  EcT_lo[c * NN + r] = l;
    splitf(E_row[i], h, l);
    Er_hi[i] = h;            Er_lo[i] = l;
}

// ---------------------------------------------------------------------------
// GEMM: C = A(MxK) * B(KxN), all dims 512, batch via blockIdx.z.
// pass 0: A = x (f32, batched, shift-on-read if inverse), B = EcT (bf16 split),
//         C -> tT hi/lo (split bf16, TRANSPOSED [n][m], batched)
// pass 1: A = Er (bf16 split),  B = tT (bf16 split, [n][h], batched),
//         C -> f32 out (batched) with freqshift + 1/(H*W) if forward
// ---------------------------------------------------------------------------
__global__ __launch_bounds__(256)
void fht2d_gemm_bf16x3(const float* __restrict__ Ax,
                       const unsigned short* __restrict__ Ah,
                       const unsigned short* __restrict__ Al,
                       const unsigned short* __restrict__ Bh,
                       const unsigned short* __restrict__ Bl,
                       unsigned short* __restrict__ Chi,
                       unsigned short* __restrict__ Clo,
                       float* __restrict__ Cf,
                       const int* __restrict__ invp,
                       int pass)
{
    const int inv    = invp[0];
    const int shiftA = (pass == 0 && inv)  ? (NN / 2) : 0;
    const int shiftC = (pass == 1 && !inv) ? (NN / 2) : 0;
    const float scale = (pass == 1 && !inv)
                      ? (1.0f / ((float)NN * (float)NN)) : 1.0f;

    const long long img = (long long)NN * NN;
    const int bz = blockIdx.z;
    if (pass == 0) { Ax += bz * img; Chi += bz * img; Clo += bz * img; }
    else           { Bh += bz * img; Bl += bz * img; Cf  += bz * img; }

    const int tile_m = blockIdx.y * TILE;
    const int tile_n = blockIdx.x * TILE;

    __shared__ __align__(16) unsigned short As_hi[2][TILE * PITCH];
    __shared__ __align__(16) unsigned short As_lo[2][TILE * PITCH];
    __shared__ __align__(16) unsigned short Bs_hi[2][TILE * PITCH];  // [n][k]
    __shared__ __align__(16) unsigned short Bs_lo[2][TILE * PITCH];

    const int tid   = threadIdx.x;
    const int lane  = tid & 31;
    const int wave  = tid >> 5;          // 8 waves
    const int wm    = (wave & 1) * 64;
    const int wn    = (wave >> 1) * 32;
    const int lhalf = lane & 15;
    const int lsel  = lane >> 4;         // 0 | 1

    // async copy of a TILE x TK bf16 tile (src row pitch 512): 8 rows/instr/wave
    auto copy_tile = [&](unsigned short* ldsdst, const unsigned short* src,
                         int grow0, int kk) {
        int r  = wave * 16 + (lane >> 2);
        int s8 = (lane & 3) * 8;
        #pragma unroll
        for (int h = 0; h < 2; ++h) {
            int rr = r + h * 8;
            unsigned off = (unsigned)(((grow0 + rr) * NN + kk + s8) * 2);
            unsigned lds = (unsigned)(unsigned long long)&ldsdst[rr * PITCH + s8];
            async_b128(lds, src, off);
        }
    };

    v4f xreg[4];  // pass-0 A prefetch registers
    auto issue_prefetch = [&](int kk, int buf) {
        copy_tile(Bs_hi[buf], Bh, tile_n, kk);
        copy_tile(Bs_lo[buf], Bl, tile_n, kk);
        if (pass == 1) {
            copy_tile(As_hi[buf], Ah, tile_m, kk);
            copy_tile(As_lo[buf], Al, tile_m, kk);
        } else {
            #pragma unroll
            for (int i = 0; i < 4; ++i) {
                int idx = tid + i * 256;          // 1024 float4 = 128x32 tile
                int r   = idx >> 3;
                int k4  = (idx & 7) << 2;
                int gr  = (tile_m + r + shiftA) & (NN - 1);
                int gk  = (kk + k4 + shiftA) & (NN - 1);  // shift never splits a float4
                xreg[i] = *reinterpret_cast<const v4f*>(Ax + (size_t)gr * NN + gk);
            }
        }
    };
    // pass 0: convert prefetched x regs -> split bf16, packed uint2 LDS stores
    auto commit_x = [&](int buf) {
        #pragma unroll
        for (int i = 0; i < 4; ++i) {
            int idx = tid + i * 256;
            int r   = idx >> 3;
            int k4  = (idx & 7) << 2;
            unsigned short h0,l0,h1,l1,h2,l2,h3,l3;
            splitf(xreg[i][0], h0, l0); splitf(xreg[i][1], h1, l1);
            splitf(xreg[i][2], h2, l2); splitf(xreg[i][3], h3, l3);
            uint2 hp = make_uint2((unsigned)h0 | ((unsigned)h1 << 16),
                                  (unsigned)h2 | ((unsigned)h3 << 16));
            uint2 lp = make_uint2((unsigned)l0 | ((unsigned)l1 << 16),
                                  (unsigned)l2 | ((unsigned)l3 << 16));
            *reinterpret_cast<uint2*>(&As_hi[buf][r * PITCH + k4]) = hp;
            *reinterpret_cast<uint2*>(&As_lo[buf][r * PITCH + k4]) = lp;
        }
    };

    const v8f vzero = {0.f,0.f,0.f,0.f,0.f,0.f,0.f,0.f};
    v8f acc[4][2];
    #pragma unroll
    for (int i = 0; i < 4; ++i)
        #pragma unroll
        for (int j = 0; j < 2; ++j)
            acc[i][j] = vzero;

    issue_prefetch(0, 0);

    // ONE barrier per k-step. Hazard chain:
    //  - commit_x(it)/async(it->buf^1) overwrite data last READ in it-2 / it-1,
    //    and those DS reads completed at the previous barrier's fence.
    //  - all writes into buf (other waves' commits + their asyncs, each wave
    //    waits its own ASYNCcnt/DScnt pre-barrier) are visible after barrier.
    for (int it = 0; it < KSTEPS; ++it) {
        const int buf = it & 1;
        if (pass == 0) commit_x(buf);        // readers of As[buf] done 1 barrier ago
        wait_async0();                        // this wave's copies into buf complete
        __syncthreads();                      // whole tile visible to all waves
        if (it + 1 < KSTEPS) issue_prefetch((it + 1) * TK, buf ^ 1);  // overlap wmma

        // B fragments: lanes 0-15 hold K0-15, lanes 16-31 hold K16-31
        v16bf bh[2], bl[2];
        #pragma unroll
        for (int fn = 0; fn < 2; ++fn) {
            int col = wn + fn * 16 + lhalf;
            int ks  = lsel * 16;
            v8bf c0 = *reinterpret_cast<const v8bf*>(&Bs_hi[buf][col * PITCH + ks]);
            v8bf c1 = *reinterpret_cast<const v8bf*>(&Bs_hi[buf][col * PITCH + ks + 8]);
            bh[fn] = __builtin_shufflevector(c0, c1, 0,1,2,3,4,5,6,7,8,9,10,11,12,13,14,15);
            v8bf d0 = *reinterpret_cast<const v8bf*>(&Bs_lo[buf][col * PITCH + ks]);
            v8bf d1 = *reinterpret_cast<const v8bf*>(&Bs_lo[buf][col * PITCH + ks + 8]);
            bl[fn] = __builtin_shufflevector(d0, d1, 0,1,2,3,4,5,6,7,8,9,10,11,12,13,14,15);
        }
        // A fragments: lanes 0-15 K0-15, lanes 16-31 K16-31
        #pragma unroll
        for (int fm = 0; fm < 4; ++fm) {
            int row = wm + fm * 16 + lhalf;
            int k0  = lsel * 16;
            v8bf a0 = *reinterpret_cast<const v8bf*>(&As_hi[buf][row * PITCH + k0]);
            v8bf a1 = *reinterpret_cast<const v8bf*>(&As_hi[buf][row * PITCH + k0 + 8]);
            v16bf ah = __builtin_shufflevector(a0, a1, 0,1,2,3,4,5,6,7,8,9,10,11,12,13,14,15);
            v8bf e0 = *reinterpret_cast<const v8bf*>(&As_lo[buf][row * PITCH + k0]);
            v8bf e1 = *reinterpret_cast<const v8bf*>(&As_lo[buf][row * PITCH + k0 + 8]);
            v16bf al = __builtin_shufflevector(e0, e1, 0,1,2,3,4,5,6,7,8,9,10,11,12,13,14,15);
            #pragma unroll
            for (int fn = 0; fn < 2; ++fn) {
                acc[fm][fn] = __builtin_amdgcn_wmma_f32_16x16x32_bf16(
                    false, ah, false, bh[fn], (short)0, acc[fm][fn], false, false);
                acc[fm][fn] = __builtin_amdgcn_wmma_f32_16x16x32_bf16(
                    false, ah, false, bl[fn], (short)0, acc[fm][fn], false, false);
                acc[fm][fn] = __builtin_amdgcn_wmma_f32_16x16x32_bf16(
                    false, al, false, bh[fn], (short)0, acc[fm][fn], false, false);
            }
        }
    }

    // D layout: lanes 0-15 M=r, lanes 16-31 M=8+r, N=lane&15
    if (pass == 0) {
        // store split bf16 TRANSPOSED: tT[n][m], 8 consecutive m per lane = b128
        #pragma unroll
        for (int fm = 0; fm < 4; ++fm) {
            #pragma unroll
            for (int fn = 0; fn < 2; ++fn) {
                int nn2 = tile_n + wn + fn * 16 + lhalf;
                int m0  = tile_m + wm + fm * 16 + lsel * 8;
                unsigned hq[4], lq[4];
                #pragma unroll
                for (int j = 0; j < 4; ++j) {
                    unsigned short h0,l0,h1,l1;
                    splitf(acc[fm][fn][2*j],     h0, l0);
                    splitf(acc[fm][fn][2*j + 1], h1, l1);
                    hq[j] = (unsigned)h0 | ((unsigned)h1 << 16);
                    lq[j] = (unsigned)l0 | ((unsigned)l1 << 16);
                }
                *reinterpret_cast<uint4*>(&Chi[(size_t)nn2 * NN + m0]) =
                    make_uint4(hq[0], hq[1], hq[2], hq[3]);
                *reinterpret_cast<uint4*>(&Clo[(size_t)nn2 * NN + m0]) =
                    make_uint4(lq[0], lq[1], lq[2], lq[3]);
            }
        }
    } else {
        #pragma unroll
        for (int fm = 0; fm < 4; ++fm) {
            #pragma unroll
            for (int fn = 0; fn < 2; ++fn) {
                #pragma unroll
                for (int r = 0; r < 8; ++r) {
                    int m = (tile_m + wm + fm * 16 + lsel * 8 + r + shiftC) & (NN - 1);
                    int n = (tile_n + wn + fn * 16 + lhalf + shiftC) & (NN - 1);
                    Cf[(size_t)m * NN + n] = acc[fm][fn][r] * scale;
                }
            }
        }
    }
}

extern "C" void kernel_launch(void* const* d_in, const int* in_sizes, int n_in,
                              void* d_out, int out_size, void* d_ws, size_t ws_size,
                              hipStream_t stream) {
    const float* x     = (const float*)d_in[0];   // [48, 512, 512]
    const float* E_col = (const float*)d_in[1];   // [512, 512]
    const float* E_row = (const float*)d_in[2];   // [512, 512]
    const int*   inv   = (const int*)d_in[3];     // scalar flag (device)

    const long long IMG = (long long)NN * NN;     // 262144
    unsigned short* tT_hi  = (unsigned short*)d_ws;          // 48*IMG bf16
    unsigned short* tT_lo  = tT_hi  + 48LL * IMG;
    unsigned short* EcT_hi = tT_lo  + 48LL * IMG;
    unsigned short* EcT_lo = EcT_hi + IMG;
    unsigned short* Er_hi  = EcT_lo + IMG;
    unsigned short* Er_lo  = Er_hi  + IMG;       // total = 100 * IMG * 2 = 52.4 MB

    fht2d_split_E<<<dim3((unsigned)(IMG / 256)), dim3(256), 0, stream>>>(
        E_col, E_row, EcT_hi, EcT_lo, Er_hi, Er_lo);

    dim3 grid(NN / TILE, NN / TILE, 48), block(256);
    // pass 0: tT[bc] = (x[bc] * E_col)^T  as split bf16
    fht2d_gemm_bf16x3<<<grid, block, 0, stream>>>(
        x, nullptr, nullptr, EcT_hi, EcT_lo, tT_hi, tT_lo, nullptr, inv, 0);
    // pass 1: out[bc] = E_row * t[bc]  (+ freqshift & 1/(H*W) if forward)
    fht2d_gemm_bf16x3<<<grid, block, 0, stream>>>(
        nullptr, Er_hi, Er_lo, tT_hi, tT_lo, nullptr, nullptr,
        (float*)d_out, inv, 1);
}
